// GatedCausalSelfAttention_20572893348704
// MI455X (gfx1250) — compile-verified, matrix operand order checked by reading, not run on hardware
//
#include <hip/hip_runtime.h>
#include <hip/hip_bf16.h>

#define DEV __device__ __forceinline__

typedef __attribute__((ext_vector_type(16))) __bf16         v16bf;
typedef __attribute__((ext_vector_type(8)))  float          v8f;
typedef __attribute__((ext_vector_type(8)))  unsigned short us8;

struct us16pair { us8 lo, hi; };
static_assert(sizeof(us16pair) == sizeof(v16bf), "frag size");

// Problem constants (B,T,D)=(2,2048,2048), H=16,G=4,HD=128, R=4
constexpr int Tc  = 2048;
constexpr int Dc  = 2048;
constexpr int Hc  = 16;
constexpr int Gc  = 4;
constexpr int HDc = 128;
constexpr int BT  = 2 * Tc;                         // 4096 rows
constexpr int NQKV = 2 * Hc * HDc + 2 * Gc * HDc;   // 5120
constexpr float ATT_SCALE = 0.08838834764831845f;   // 1/sqrt(128)
constexpr float NEG_BIG = -3.0e38f;

DEV unsigned short f2bf(float f) {
  union { float f; unsigned u; } v; v.f = f;
  unsigned r = v.u + 0x7FFFu + ((v.u >> 16) & 1u);  // RNE
  return (unsigned short)(r >> 16);
}

DEV v16bf mk_frag(us8 lo, us8 hi) {
  us16pair p{lo, hi};
  return __builtin_bit_cast(v16bf, p);
}

// 16-bit WMMA operand fragment: lo = elems [base..base+7], hi = [base+16..base+23]
// with base = 8*(lane>>4); p must be 16B aligned.
DEV v16bf load_frag16(const unsigned short* p) {
  return mk_frag(*(const us8*)p, *(const us8*)(p + 16));
}

DEV v8f wmma_bf16(v16bf a, v16bf b, v8f c) {
  return __builtin_amdgcn_wmma_f32_16x16x32_bf16(false, a, false, b, (short)0, c,
                                                 false, false);
}

// ---------------------------------------------------------------- converts
__global__ void cvt_bf16_kernel(const float* __restrict__ in,
                                unsigned short* __restrict__ out, size_t n) {
  size_t stride = (size_t)gridDim.x * blockDim.x;
  for (size_t i = (size_t)blockIdx.x * blockDim.x + threadIdx.x; i < n; i += stride)
    out[i] = f2bf(in[i]);
}

__global__ void pack_wqkv_kernel(const float* __restrict__ wq,
                                 const float* __restrict__ wk,
                                 const float* __restrict__ wv,
                                 unsigned short* __restrict__ out) {
  const size_t n = (size_t)Dc * NQKV;
  size_t stride = (size_t)gridDim.x * blockDim.x;
  for (size_t i = (size_t)blockIdx.x * blockDim.x + threadIdx.x; i < n; i += stride) {
    size_t d = i / NQKV;
    int col = (int)(i % NQKV);
    float v;
    if (col < 4096)      v = wq[d * 4096 + col];
    else if (col < 4608) v = wk[d * 512 + (col - 4096)];
    else                 v = wv[d * 512 + (col - 4608)];
    out[i] = f2bf(v);
  }
}

// ---------------------------------------------------------------- GEMM
// C[M,N] fp32 = A[M,K] bf16 (row major) * B[K,N] bf16 (row major)
// block = 256 threads (8 waves), tile 128x128, K-step 32, double-buffered LDS.
__global__ __launch_bounds__(256) void gemm_bf16_kernel(
    const unsigned short* __restrict__ A, const unsigned short* __restrict__ B,
    float* __restrict__ C, int M, int N, int K) {
  __shared__ __align__(16) unsigned short As[2][128 * 32];  // [row][k]
  __shared__ __align__(16) unsigned short Bt[2][128 * 32];  // [n][k] transposed

  const int m0 = blockIdx.y * 128;
  const int n0 = blockIdx.x * 128;
  const int tid = threadIdx.x;
  const int wid = tid >> 5;
  const int lane = tid & 31;
  const int lrow = lane & 15;
  const int lhalf = lane >> 4;

  const v8f vzero = {0.f, 0.f, 0.f, 0.f, 0.f, 0.f, 0.f, 0.f};
  v8f acc[8];
#pragma unroll
  for (int i = 0; i < 8; ++i) acc[i] = vzero;

  const int arow = tid >> 1;        // 0..127
  const int acol = (tid & 1) * 16;  // 0 / 16
  const int brow = tid >> 3;        // 0..31 (k)
  const int bcol = (tid & 7) * 16;  // 0..112 (n)
  const int KT = K >> 5;

  auto stage = [&](int buf, int k0) {
    const us8* asrc = (const us8*)(A + (size_t)(m0 + arow) * K + k0 + acol);
    us8 a0 = asrc[0], a1 = asrc[1];
    *(us8*)&As[buf][arow * 32 + acol]     = a0;
    *(us8*)&As[buf][arow * 32 + acol + 8] = a1;
    const us8* bsrc = (const us8*)(B + (size_t)(k0 + brow) * N + n0 + bcol);
    us8 b0 = bsrc[0], b1 = bsrc[1];
#pragma unroll
    for (int j = 0; j < 8; ++j) Bt[buf][(bcol + j) * 32 + brow] = b0[j];
#pragma unroll
    for (int j = 0; j < 8; ++j) Bt[buf][(bcol + 8 + j) * 32 + brow] = b1[j];
    if (k0 + 64 < K) {
      __builtin_prefetch((const void*)(A + (size_t)(m0 + arow) * K + k0 + 64 + acol), 0, 1);
      __builtin_prefetch((const void*)(B + (size_t)(k0 + 64 + brow) * N + n0 + bcol), 0, 1);
    }
  };

  stage(0, 0);
  __syncthreads();

  for (int kt = 0; kt < KT; ++kt) {
    const int cur = kt & 1;
    if (kt + 1 < KT) stage(cur ^ 1, (kt + 1) * 32);

    // fragments first, WMMAs after — lets ds latency hide under wmma issue
    v16bf af = load_frag16(&As[cur][(wid * 16 + lrow) * 32 + 8 * lhalf]);
    v16bf bfr[8];
#pragma unroll
    for (int nc = 0; nc < 8; ++nc)
      bfr[nc] = load_frag16(&Bt[cur][(nc * 16 + lrow) * 32 + 8 * lhalf]);
#pragma unroll
    for (int nc = 0; nc < 8; ++nc) acc[nc] = wmma_bf16(af, bfr[nc], acc[nc]);

    __syncthreads();
  }

  // C layout: lane L -> col L%16, rows r + 8*(L>>4)
#pragma unroll
  for (int nc = 0; nc < 8; ++nc) {
#pragma unroll
    for (int r = 0; r < 8; ++r) {
      int row = m0 + wid * 16 + lhalf * 8 + r;
      int col = n0 + nc * 16 + lrow;
      C[(size_t)row * N + col] = acc[nc][r];
    }
  }
}

// ---------------------------------------------------------------- post-projection
DEV void norm_rope_store(const float* __restrict__ src, const float* __restrict__ w,
                         const float* __restrict__ ct, const float* __restrict__ st,
                         unsigned short* __restrict__ dst, int lane) {
  float v0 = src[lane], v1 = src[lane + 32], v2 = src[lane + 64], v3 = src[lane + 96];
  float ss = v0 * v0 + v1 * v1 + v2 * v2 + v3 * v3;
#pragma unroll
  for (int m = 16; m >= 1; m >>= 1) ss += __shfl_xor(ss, m, 32);
  float sc = rsqrtf(ss * (1.0f / 128.0f) + 1e-6f);
  v0 *= sc * w[lane];      v1 *= sc * w[lane + 32];
  v2 *= sc * w[lane + 64]; v3 *= sc * w[lane + 96];
  // rotate_half pairing: (d, d+64) owned in-lane as (v0,v2), (v1,v3)
  dst[lane]      = f2bf(v0 * ct[lane]      - v2 * st[lane]);
  dst[lane + 32] = f2bf(v1 * ct[lane + 32] - v3 * st[lane + 32]);
  dst[lane + 64] = f2bf(v2 * ct[lane + 64] + v0 * st[lane + 64]);
  dst[lane + 96] = f2bf(v3 * ct[lane + 96] + v1 * st[lane + 96]);
}

__global__ __launch_bounds__(256) void postproc_kernel(
    const float* __restrict__ qkv, const float* __restrict__ qnw,
    const float* __restrict__ knw, const float* __restrict__ cosT,
    const float* __restrict__ sinT, unsigned short* __restrict__ qb,
    unsigned short* __restrict__ kb, unsigned short* __restrict__ vb,
    float* __restrict__ gateS) {
  const int row = blockIdx.x;  // b*T + t
  const int b = row / Tc, t = row % Tc;
  const int wid = threadIdx.x >> 5, lane = threadIdx.x & 31;
  const float* base = qkv + (size_t)row * NQKV;
  const float* ct = cosT + (size_t)t * HDc;
  const float* st = sinT + (size_t)t * HDc;

  for (int seg = wid; seg < 24; seg += 8) {
    if (seg < 16) {  // q head + its gate
      int h = seg;
      norm_rope_store(base + h * 256, qnw, ct, st,
                      qb + (((size_t)(b * Hc + h)) * Tc + t) * HDc, lane);
      const float* gsrc = base + h * 256 + 128;
      float* gdst = gateS + (size_t)row * (Hc * HDc) + h * HDc;
#pragma unroll
      for (int i = 0; i < 4; ++i) {
        float gv = gsrc[lane + 32 * i];
        gdst[lane + 32 * i] = 1.0f / (1.0f + __expf(-gv));
      }
    } else if (seg < 20) {  // k group
      int g = seg - 16;
      norm_rope_store(base + 4096 + g * 128, knw, ct, st,
                      kb + (((size_t)(b * Gc + g)) * Tc + t) * HDc, lane);
    } else {  // v group -> bf16
      int g = seg - 20;
      const float* src = base + 4608 + g * 128;
      unsigned short* dst = vb + (((size_t)(b * Gc + g)) * Tc + t) * HDc;
#pragma unroll
      for (int i = 0; i < 4; ++i) dst[lane + 32 * i] = f2bf(src[lane + 32 * i]);
    }
  }
}

// ---------------------------------------------------------------- flash attention
// Block = 8 waves = 8 consecutive 16-row q-tiles of ONE (b,h): K/V tiles are
// staged cooperatively in LDS (K row-major, V transposed) with b128 traffic.
// Uniform trip count across the block; causal mask makes overshoot tiles inert.
__global__ __launch_bounds__(256) void attn_kernel(
    const unsigned short* __restrict__ qb, const unsigned short* __restrict__ kbuf,
    const unsigned short* __restrict__ vbuf, const float* __restrict__ gateS,
    unsigned short* __restrict__ gated) {
  __shared__ __align__(16) unsigned short Ks[32 * 128];   // [key][hd]
  __shared__ __align__(16) unsigned short Vt[128 * 32];   // [hd][key]
  __shared__ __align__(16) unsigned short Pl[8][16 * 32]; // per-wave P bounce

  const int tid = threadIdx.x;
  const int wid = tid >> 5, lane = tid & 31;
  const int lrow = lane & 15, lhalf = lane >> 4;
  const int task0 = blockIdx.x * 8;        // 8 q-tiles, same (b,h)
  const int bh = task0 >> 7;
  const int qt0 = task0 & 127;
  const int qt = qt0 + wid;
  const int b = bh >> 4, h = bh & 15;
  const int g = h >> 2;                    // R = 4
  const int qrow0 = qt * 16;

  const unsigned short* qhead = qb + ((size_t)(b * Hc + h) * Tc + qrow0) * HDc;
  const unsigned short* kh = kbuf + (size_t)(b * Gc + g) * Tc * HDc;
  const unsigned short* vh = vbuf + (size_t)(b * Gc + g) * Tc * HDc;
  unsigned short* pl = Pl[wid];

  // resident Q fragments: 16x128 = 4 chunks of 16x32
  v16bf qf[4];
#pragma unroll
  for (int c = 0; c < 4; ++c)
    qf[c] = load_frag16(qhead + (size_t)lrow * HDc + c * 32 + 8 * lhalf);

  const v8f vzero = {0.f, 0.f, 0.f, 0.f, 0.f, 0.f, 0.f, 0.f};
  v8f o[8];
#pragma unroll
  for (int i = 0; i < 8; ++i) o[i] = vzero;
  float mi[8], li[8];
#pragma unroll
  for (int r = 0; r < 8; ++r) { mi[r] = NEG_BIG; li[r] = 0.f; }

  const int srow = tid >> 3;        // 0..31 staged key row
  const int scol = (tid & 7) * 16;  // 0..112 hd col
  const int nst = ((qt0 + 8) * 16 + 31) >> 5;  // uniform over block

  for (int stp = 0; stp < nst; ++stp) {
    const int s0 = stp * 32;
    // ---- cooperative staging: K row-major, V transposed
    {
      const us8* ksrc = (const us8*)(kh + (size_t)(s0 + srow) * HDc + scol);
      us8 ka = ksrc[0], kb2 = ksrc[1];
      *(us8*)&Ks[srow * HDc + scol]     = ka;
      *(us8*)&Ks[srow * HDc + scol + 8] = kb2;
      const us8* vsrc = (const us8*)(vh + (size_t)(s0 + srow) * HDc + scol);
      us8 va = vsrc[0], vb2 = vsrc[1];
#pragma unroll
      for (int j = 0; j < 8; ++j) Vt[(scol + j) * 32 + srow] = va[j];
#pragma unroll
      for (int j = 0; j < 8; ++j) Vt[(scol + 8 + j) * 32 + srow] = vb2[j];
    }
    __syncthreads();

    // ---- scores S[16x32] = Q @ K^T (two 16x16 C-fragments)
    v8f sf0 = vzero, sf1 = vzero;
#pragma unroll
    for (int c = 0; c < 4; ++c) {
      sf0 = wmma_bf16(qf[c], load_frag16(&Ks[lrow * HDc + c * 32 + 8 * lhalf]), sf0);
      sf1 = wmma_bf16(qf[c], load_frag16(&Ks[(16 + lrow) * HDc + c * 32 + 8 * lhalf]), sf1);
    }

    // ---- scale + causal mask + online softmax
    float tm[8];
#pragma unroll
    for (int r = 0; r < 8; ++r) {
      float a0 = sf0[r] * ATT_SCALE, a1 = sf1[r] * ATT_SCALE;
      int rowg = qrow0 + lhalf * 8 + r;
      if (s0 + lrow > rowg)      a0 = NEG_BIG;
      if (s0 + 16 + lrow > rowg) a1 = NEG_BIG;
      sf0[r] = a0; sf1[r] = a1;
      tm[r] = fmaxf(a0, a1);
    }
#pragma unroll
    for (int r = 0; r < 8; ++r) {
#pragma unroll
      for (int m = 8; m >= 1; m >>= 1) tm[r] = fmaxf(tm[r], __shfl_xor(tm[r], m, 32));
    }
    float alpha[8];
#pragma unroll
    for (int r = 0; r < 8; ++r) {
      float nm = fmaxf(mi[r], tm[r]);
      alpha[r] = __expf(mi[r] - nm);
      mi[r] = nm;
      float p0 = __expf(sf0[r] - nm), p1 = __expf(sf1[r] - nm);
      sf0[r] = p0; sf1[r] = p1;
      float psum = p0 + p1;
#pragma unroll
      for (int m = 8; m >= 1; m >>= 1) psum += __shfl_xor(psum, m, 32);
      li[r] = li[r] * alpha[r] + psum;
    }
#pragma unroll
    for (int nc = 0; nc < 8; ++nc)
#pragma unroll
      for (int r = 0; r < 8; ++r) o[nc][r] = o[nc][r] * alpha[r];

    // ---- P (C-frag) -> A-frag via per-wave LDS bounce
#pragma unroll
    for (int r = 0; r < 8; ++r) {
      pl[(lhalf * 8 + r) * 32 + lrow]      = f2bf(sf0[r]);
      pl[(lhalf * 8 + r) * 32 + 16 + lrow] = f2bf(sf1[r]);
    }
    asm volatile("s_wait_dscnt 0" ::: "memory");
    v16bf pf = load_frag16(&pl[lrow * 32 + 8 * lhalf]);

    // ---- O += P @ V from transposed LDS tile
    v16bf vfr[8];
#pragma unroll
    for (int nc = 0; nc < 8; ++nc)
      vfr[nc] = load_frag16(&Vt[(nc * 16 + lrow) * 32 + 8 * lhalf]);
#pragma unroll
    for (int nc = 0; nc < 8; ++nc) o[nc] = wmma_bf16(pf, vfr[nc], o[nc]);

    __syncthreads();  // protect Ks/Vt for next tile
  }

  // ---- epilogue: O/l * sigmoid(gate) -> bf16 [B*T, H*HD]
#pragma unroll
  for (int nc = 0; nc < 8; ++nc) {
#pragma unroll
    for (int r = 0; r < 8; ++r) {
      int trow = qrow0 + lhalf * 8 + r;
      size_t idx = ((size_t)b * Tc + trow) * (size_t)(Hc * HDc) + h * HDc + nc * 16 + lrow;
      float val = (o[nc][r] / li[r]) * gateS[idx];
      gated[idx] = f2bf(val);
    }
  }
}

// ---------------------------------------------------------------- host
extern "C" void kernel_launch(void* const* d_in, const int* in_sizes, int n_in,
                              void* d_out, int out_size, void* d_ws, size_t ws_size,
                              hipStream_t stream) {
  (void)in_sizes; (void)n_in; (void)out_size; (void)ws_size;
  const float* x    = (const float*)d_in[0];
  const float* w_q  = (const float*)d_in[1];
  const float* w_k  = (const float*)d_in[2];
  const float* w_v  = (const float*)d_in[3];
  const float* w_o  = (const float*)d_in[4];
  const float* qnw  = (const float*)d_in[5];
  const float* knw  = (const float*)d_in[6];
  const float* cosT = (const float*)d_in[7];
  const float* sinT = (const float*)d_in[8];
  float* out = (float*)d_out;

  char* p = (char*)d_ws;
  auto take = [&](size_t bytes) {
    char* r = p;
    p += (bytes + 255) & ~(size_t)255;
    return r;
  };
  unsigned short* x_bf    = (unsigned short*)take((size_t)BT * Dc * 2);
  unsigned short* wqkv_bf = (unsigned short*)take((size_t)Dc * NQKV * 2);
  unsigned short* wo_bf   = (unsigned short*)take((size_t)Dc * Dc * 2);
  float*          qkv     = (float*)take((size_t)BT * NQKV * 4);
  unsigned short* q_bf    = (unsigned short*)take((size_t)2 * Hc * Tc * HDc * 2);
  unsigned short* k_bf    = (unsigned short*)take((size_t)2 * Gc * Tc * HDc * 2);
  unsigned short* v_bf    = (unsigned short*)take((size_t)2 * Gc * Tc * HDc * 2);
  float*          gateS   = (float*)take((size_t)BT * Hc * HDc * 4);
  unsigned short* gated   = (unsigned short*)take((size_t)BT * Hc * HDc * 2);

  cvt_bf16_kernel<<<1024, 256, 0, stream>>>(x, x_bf, (size_t)BT * Dc);
  cvt_bf16_kernel<<<1024, 256, 0, stream>>>(w_o, wo_bf, (size_t)Dc * Dc);
  pack_wqkv_kernel<<<1024, 256, 0, stream>>>(w_q, w_k, w_v, wqkv_bf);

  // QKV projection: [4096,2048] x [2048,5120] -> fp32
  gemm_bf16_kernel<<<dim3(NQKV / 128, BT / 128), 256, 0, stream>>>(
      x_bf, wqkv_bf, qkv, BT, NQKV, Dc);

  postproc_kernel<<<BT, 256, 0, stream>>>(qkv, qnw, knw, cosT, sinT, q_bf, k_bf,
                                          v_bf, gateS);

  // B*H*(T/16) = 4096 wave-tasks, 8 per block sharing one (b,h)
  attn_kernel<<<512, 256, 0, stream>>>(q_bf, k_bf, v_bf, gateS, gated);

  // output projection: [4096,2048] x [2048,2048] -> fp32 out
  gemm_bf16_kernel<<<dim3(Dc / 128, BT / 128), 256, 0, stream>>>(
      gated, wo_bf, out, BT, Dc, Dc);
}
